// HedgehogAttention_13932873908885
// MI455X (gfx1250) — matrix operand order repaired and checked by reading, not compiled, not run on hardware
//
#include <hip/hip_runtime.h>
#include <hip/hip_bf16.h>

// ---------------------------------------------------------------------------
// Hedgehog attention for MI455X (gfx1250): fp16 WMMA flash-attention pipeline.
// v2: 32 queries per wave (2 tiles) so each K/V fragment feeds 2 WMMAs, and
// all fragment loads of a step are hoisted ahead of the WMMA burst (single
// load clause + one s_wait instead of per-WMMA waits).
// ---------------------------------------------------------------------------

typedef __attribute__((ext_vector_type(16))) _Float16 v16h;
typedef __attribute__((ext_vector_type(8)))  _Float16 v8h;
typedef __attribute__((ext_vector_type(8)))  float    v8f;

constexpr int NB    = 4;      // batch
constexpr int SLEN  = 2048;   // sequence
constexpr int HEADS = 16;
constexpr int HDIM  = 64;
constexpr int FDIM  = 128;    // expanded feature dim (2*HDIM)
constexpr int EMBED = 1024;
constexpr int OUTD  = HEADS * FDIM; // 2048 == final GEMM K
constexpr float SCALE = 0.03125f;   // 1/sqrt(EMBED)

// --- WMMA fragment loaders (contiguous k order, not the ISA table order, wave32, 16-bit operands) ----
// A (16x32, MxK): lane holds row M=lane%16; halves [0..15] = K in
// [o..o+15], o = (lane>=16 ? 16 : 0).
// B (32x16, KxN): lane holds col N=lane%16; halves [0..15] = K in
// [g*16 .. g*16+15], g = lane>>4.  All runs are 16-byte aligned.

__device__ __forceinline__ v16h make_frag(v8h lo, v8h hi) {
  v16h r;
#pragma unroll
  for (int i = 0; i < 8; ++i) { r[i] = lo[i]; r[i + 8] = hi[i]; }
  return r;
}

__device__ __forceinline__ v16h load_fragA(const _Float16* row_ptr, int lane) {
  const int o = (lane & 16) ? 16 : 0;
  v8h lo = *(const v8h*)(row_ptr + o);
  v8h hi = *(const v8h*)(row_ptr + o + 8);
  return make_frag(lo, hi);
}

__device__ __forceinline__ v16h load_fragB(const _Float16* p16) {
  v8h lo = *(const v8h*)(p16);
  v8h hi = *(const v8h*)(p16 + 8);
  return make_frag(lo, hi);
}

#define WMMA_F16(A, B, C) \
  __builtin_amdgcn_wmma_f32_16x16x32_f16(false, (A), false, (B), (short)0, (C), false, false)

// --------------------------- Stage 1: feature map ---------------------------
__global__ __launch_bounds__(64)
void hh_feature_map(const float* __restrict__ xq, const float* __restrict__ xk,
                    const float* __restrict__ xv,
                    const float* __restrict__ fmw, const float* __restrict__ fmb,
                    _Float16* __restrict__ qf, _Float16* __restrict__ kf,
                    _Float16* __restrict__ vt) {
  __shared__ float xs[HDIM];
  const int e = threadIdx.x;       // 0..63 output feature
  const int t = blockIdx.x;        // token 0..N*S-1
  const int h = blockIdx.y;        // head
  const int which = blockIdx.z;    // 0=q 1=k 2=v
  const float* x = (which == 0) ? xq : (which == 1) ? xk : xv;
  xs[e] = x[(size_t)t * EMBED + h * HDIM + e];
  __syncthreads();
  const float* w = fmw + e * HDIM;
  float z = fmb[e];
#pragma unroll
  for (int d = 0; d < HDIM; ++d) z = fmaf(xs[d], w[d], z);
  const float ep = __expf(z), en = __expf(-z);
  const int n = t / SLEN, s = t % SLEN;
  const int nh = n * HEADS + h;
  if (which == 2) {
    vt[((size_t)nh * FDIM + e) * SLEN + s]        = (_Float16)ep;
    vt[((size_t)nh * FDIM + HDIM + e) * SLEN + s] = (_Float16)en;
  } else {
    _Float16* o = (which == 0) ? qf : kf;
    const size_t base = ((size_t)nh * SLEN + s) * FDIM;
    o[base + e]        = (_Float16)ep;
    o[base + HDIM + e] = (_Float16)en;
  }
}

__global__ __launch_bounds__(256)
void hh_cvt_f16(const float* __restrict__ w, _Float16* __restrict__ w16, int n) {
  const int i = blockIdx.x * 256 + threadIdx.x;
  if (i < n) w16[i] = (_Float16)w[i];
}

// ----------------------- Stage 2: flash attention ---------------------------
// One wave per (nh, 32-query block = 2 WMMA row-tiles). Per 32-key step:
//   - 8 K B-fragments loaded as one clause, then 16 score WMMAs (2 q-tiles)
//   - online softmax (shfl_xor row reductions within 16-lane half-groups)
//   - P re-laid out C->A through LDS (both tiles)
//   - 8 V B-fragments loaded as one clause, then 16 P*V WMMAs
__global__ __launch_bounds__(32)
void hh_attention(const _Float16* __restrict__ qf, const _Float16* __restrict__ kf,
                  const _Float16* __restrict__ vt, _Float16* __restrict__ aout) {
  __shared__ alignas(16) _Float16 ldsP[2][16 * 32];
  const int lane = threadIdx.x & 31;
  const int ln = lane & 15;
  const int hi = lane >> 4;        // half-wave group: rows 0-7 vs 8-15
  const int qt = blockIdx.x;       // 32-query block, 0..S/32-1
  const int nh = blockIdx.y;       // 0..N*H-1

  const _Float16* Q = qf + (size_t)nh * SLEN * FDIM;
  const _Float16* K = kf + (size_t)nh * SLEN * FDIM;
  const _Float16* V = vt + (size_t)nh * FDIM * SLEN;

  // Resident Q A-fragments: 2 row tiles x 4 K-chunks (128-wide feature dim).
  v16h qa[2][4];
#pragma unroll
  for (int qi = 0; qi < 2; ++qi) {
    const _Float16* qrow = Q + (size_t)(qt * 32 + qi * 16 + ln) * FDIM;
#pragma unroll
    for (int kc = 0; kc < 4; ++kc) qa[qi][kc] = load_fragA(qrow + kc * 32, lane);
  }

  v8f acc[2][8];
  float m[2][8], lsum[2][8];
#pragma unroll
  for (int qi = 0; qi < 2; ++qi)
#pragma unroll
    for (int f = 0; f < 8; ++f) {
      acc[qi][f] = (v8f){0.f, 0.f, 0.f, 0.f, 0.f, 0.f, 0.f, 0.f};
      m[qi][f] = -1e30f;
      lsum[qi][f] = 0.f;
    }

  for (int kb = 0; kb < SLEN; kb += 32) {
    const _Float16* kr0 = K + (size_t)(kb + ln) * FDIM + hi * 16;       // cols 0-15
    const _Float16* kr1 = K + (size_t)(kb + 16 + ln) * FDIM + hi * 16;  // cols 16-31
    if (kb + 32 < SLEN)  // global_prefetch_b8 of next key tile
      __builtin_prefetch(kr0 + (size_t)32 * FDIM, 0, 1);

    // ---- stage all 8 K fragments, then burst 16 score WMMAs ----
    v16h bk[8];
#pragma unroll
    for (int kc = 0; kc < 4; ++kc) {
      bk[kc]     = load_fragB(kr0 + kc * 32);
      bk[4 + kc] = load_fragB(kr1 + kc * 32);
    }
    v8f c[2][2];
#pragma unroll
    for (int qi = 0; qi < 2; ++qi)
#pragma unroll
      for (int j = 0; j < 2; ++j)
        c[qi][j] = (v8f){0.f, 0.f, 0.f, 0.f, 0.f, 0.f, 0.f, 0.f};
#pragma unroll
    for (int kc = 0; kc < 4; ++kc)
#pragma unroll
      for (int qi = 0; qi < 2; ++qi) {
        c[qi][0] = WMMA_F16(qa[qi][kc], bk[kc],     c[qi][0]);
        c[qi][1] = WMMA_F16(qa[qi][kc], bk[4 + kc], c[qi][1]);
      }

    // ---- online softmax per row (rows split across 16-lane half-groups) ----
    float p0[2][8], p1[2][8];
#pragma unroll
    for (int qi = 0; qi < 2; ++qi)
#pragma unroll
      for (int r = 0; r < 8; ++r) {
        const float s0 = c[qi][0][r] * SCALE, s1 = c[qi][1][r] * SCALE;
        float t = fmaxf(s0, s1);
#pragma unroll
        for (int mk = 1; mk < 16; mk <<= 1) t = fmaxf(t, __shfl_xor(t, mk, 32));
        const float mn = fmaxf(m[qi][r], t);
        const float corr = __expf(m[qi][r] - mn);
        m[qi][r] = mn;
        p0[qi][r] = __expf(s0 - mn);
        p1[qi][r] = __expf(s1 - mn);
        float rs = p0[qi][r] + p1[qi][r];
#pragma unroll
        for (int mk = 1; mk < 16; mk <<= 1) rs += __shfl_xor(rs, mk, 32);
        lsum[qi][r] = lsum[qi][r] * corr + rs;
#pragma unroll
        for (int f = 0; f < 8; ++f) acc[qi][f][r] *= corr;
      }

    // ---- re-layout P: C layout -> A layout via LDS (both tiles) ----
    __syncthreads();  // prior iteration's ldsP reads complete
#pragma unroll
    for (int qi = 0; qi < 2; ++qi)
#pragma unroll
      for (int r = 0; r < 8; ++r) {
        ldsP[qi][(r + 8 * hi) * 32 + ln]      = (_Float16)p0[qi][r];
        ldsP[qi][(r + 8 * hi) * 32 + 16 + ln] = (_Float16)p1[qi][r];
      }
    __syncthreads();
    v16h pa[2];
#pragma unroll
    for (int qi = 0; qi < 2; ++qi)
      pa[qi] = load_fragA((const _Float16*)&ldsP[qi][0] + ln * 32, lane);

    // ---- stage all 8 V fragments, then burst 16 P*V WMMAs ----
    v16h vb[8];
#pragma unroll
    for (int f = 0; f < 8; ++f)
      vb[f] = load_fragB(V + (size_t)(f * 16 + ln) * SLEN + kb + hi * 16);
#pragma unroll
    for (int f = 0; f < 8; ++f)
#pragma unroll
      for (int qi = 0; qi < 2; ++qi)
        acc[qi][f] = WMMA_F16(pa[qi], vb[f], acc[qi][f]);
  }

  // ---- normalize, emit f16 [token, H*128] for the projection GEMM ----
  const int n = nh / HEADS, h = nh % HEADS;
#pragma unroll
  for (int qi = 0; qi < 2; ++qi)
#pragma unroll
    for (int r = 0; r < 8; ++r) {
      const float inv = 1.f / lsum[qi][r];
      const int q = qt * 32 + qi * 16 + r + 8 * hi;
      const size_t base = ((size_t)(n * SLEN + q)) * OUTD + (size_t)h * FDIM;
#pragma unroll
      for (int f = 0; f < 8; ++f)
        aout[base + f * 16 + ln] = (_Float16)(acc[qi][f][r] * inv);
    }
}

// -------------------- Stage 3: output projection GEMM -----------------------
// out[8192 x 1024] = aout[8192 x 2048](f16) x fc_w^T(f16) + fc_b, f32 result.
// One wave: 16 rows x 128 cols (A fragment reused 8x); per K-chunk all 8 B
// fragments staged as one clause before the 8-WMMA burst.
__global__ __launch_bounds__(32)
void hh_proj(const _Float16* __restrict__ aout, const _Float16* __restrict__ w16,
             const float* __restrict__ bias, float* __restrict__ out) {
  const int lane = threadIdx.x & 31;
  const int ln = lane & 15;
  const int g = lane >> 4;
  const int bt = blockIdx.x;   // token tile (16 rows)
  const int bc = blockIdx.y;   // 128-col group (0..7)
  v8f acc[8];
#pragma unroll
  for (int j = 0; j < 8; ++j) acc[j] = (v8f){0.f, 0.f, 0.f, 0.f, 0.f, 0.f, 0.f, 0.f};

  const _Float16* arow = aout + (size_t)(bt * 16 + ln) * OUTD;
  for (int kc = 0; kc < OUTD / 32; ++kc) {
    const v16h a = load_fragA(arow + kc * 32, lane);
    v16h b[8];
#pragma unroll
    for (int j = 0; j < 8; ++j)
      b[j] = load_fragB(w16 + (size_t)((bc * 8 + j) * 16 + ln) * OUTD + kc * 32 + g * 16);
#pragma unroll
    for (int j = 0; j < 8; ++j) acc[j] = WMMA_F16(a, b[j], acc[j]);
  }
#pragma unroll
  for (int j = 0; j < 8; ++j) {
    const int col = (bc * 8 + j) * 16 + ln;
    const float bv = bias[col];
#pragma unroll
    for (int r = 0; r < 8; ++r) {
      const int row = bt * 16 + r + 8 * g;
      out[(size_t)row * EMBED + col] = acc[j][r] + bv;
    }
  }
}

// ------------------------------- launch -------------------------------------
extern "C" void kernel_launch(void* const* d_in, const int* in_sizes, int n_in,
                              void* d_out, int out_size, void* d_ws, size_t ws_size,
                              hipStream_t stream) {
  // setup_inputs order: values, keys, query, fm_w, fm_b, fc_w, fc_b (all f32)
  const float* values = (const float*)d_in[0];
  const float* keys   = (const float*)d_in[1];
  const float* query  = (const float*)d_in[2];
  const float* fm_w   = (const float*)d_in[3];
  const float* fm_b   = (const float*)d_in[4];
  const float* fc_w   = (const float*)d_in[5];
  const float* fc_b   = (const float*)d_in[6];
  float* out = (float*)d_out;

  // Workspace carve-up (f16 halves): qf | kf | vt | aout | w16  (~138 MB)
  const size_t featElems = (size_t)NB * HEADS * SLEN * FDIM;   // 16.8M each
  const size_t aoutElems = (size_t)NB * SLEN * OUTD;           // 16.8M
  _Float16* qf   = (_Float16*)d_ws;
  _Float16* kf   = qf + featElems;
  _Float16* vt   = kf + featElems;
  _Float16* aout = vt + featElems;
  _Float16* w16  = aout + aoutElems;

  // Stage 1: feature maps (q,k row-major; v transposed) + fc_w f32->f16.
  dim3 g1(NB * SLEN, HEADS, 3);
  hh_feature_map<<<g1, 64, 0, stream>>>(query, keys, values, fm_w, fm_b, qf, kf, vt);
  const int wn = EMBED * OUTD;
  hh_cvt_f16<<<(wn + 255) / 256, 256, 0, stream>>>(fc_w, w16, wn);

  // Stage 2: flash attention, one wave per (32-query block, n*h).
  dim3 g2(SLEN / 32, NB * HEADS);
  hh_attention<<<g2, 32, 0, stream>>>(qf, kf, vt, aout);

  // Stage 3: output projection, one wave per 16x128 output tile.
  dim3 g3((NB * SLEN) / 16, EMBED / 128);
  hh_proj<<<g3, 32, 0, stream>>>(aout, w16, fc_b, out);
}
